// GNNEncoder_60232621359681
// MI455X (gfx1250) — matrix operand order repaired and checked by reading, not compiled, not run on hardware
//
#include <hip/hip_runtime.h>
#include <hip/hip_bf16.h>
#include <math.h>

// ---------------------------------------------------------------------------
// Types for CDNA5 WMMA (wave32): 16x16x32 bf16 -> f32
// ---------------------------------------------------------------------------
typedef __attribute__((ext_vector_type(16))) __bf16 v16bf;
typedef __attribute__((ext_vector_type(8)))  float  v8f;
typedef __bf16 bf16_t;

#define CDIM 128
#define HDIM 4
#define DDIM 32
#define BN_EPS 1e-5f

// ---------------------------------------------------------------------------
// Small device helpers
// ---------------------------------------------------------------------------
__device__ __forceinline__ bf16_t f2bf(float f) {
  unsigned u = __float_as_uint(f);
  unsigned r = (u + 0x7fffu + ((u >> 16) & 1u)) >> 16;   // round-nearest-even
  unsigned short h = (unsigned short)r;
  bf16_t b;
  __builtin_memcpy(&b, &h, sizeof(b));
  return b;
}
__device__ __forceinline__ unsigned f2ord(float f) {
  unsigned u = __float_as_uint(f);
  return (u & 0x80000000u) ? ~u : (u | 0x80000000u);
}
__device__ __forceinline__ float ord2f(unsigned u) {
  return (u & 0x80000000u) ? __uint_as_float(u & 0x7fffffffu)
                           : __uint_as_float(~u);
}
__device__ __forceinline__ float gelu_exact(float x) {
  return 0.5f * x * (1.0f + erff(x * 0.70710678118654752f));
}

// ---------------------------------------------------------------------------
// Pack f32 W[K][N] (row-major) into bf16 B-fragments.
// B fragment (32x16 bf16, wave32): lane<16 holds col N=lane, K=0..15 as
// elements 0..15; lane>=16 holds col N=lane-16, K=16..31. Packed layout is
// fragment-contiguous: dst[((ntile*Kt + ktile)*32 + lane)*16 + e].
// ---------------------------------------------------------------------------
__global__ void pack_b_kernel(const float* __restrict__ W, bf16_t* __restrict__ dst,
                              int K, int N) {
  int t = blockIdx.x * blockDim.x + threadIdx.x;
  int Kt = K >> 5, Nt = N >> 4;
  int total = Nt * Kt * 32;
  if (t >= total) return;
  int lane = t & 31;
  int kt = (t >> 5) % Kt;
  int nt = (t >> 5) / Kt;
  int kbase = kt * 32 + ((lane >= 16) ? 16 : 0);
  int col = nt * 16 + (lane & 15);
  bf16_t* o = dst + (size_t)t * 16;
#pragma unroll
  for (int e = 0; e < 16; ++e)
    o[e] = f2bf(W[(size_t)(kbase + e) * N + col]);
}

// ---------------------------------------------------------------------------
// WMMA GEMM: out[M x (NT*16)] = A[M x K (cols acol..acol+K)] * Bpacked + bias
// Block = 256 threads = 8 waves; each wave owns a 16-row strip and NT 16x16
// f32 accumulators. Packed B (<= 32KB) is staged once per block into LDS with
// a cooperative copy, then fragments come from ds_load. All NT B fragments
// are fetched before the WMMA chain so the v_wmma issue back-to-back.
// A fragment (16x32 bf16): lane<16 -> row M=lane, K 0..15;
//                          lane>=16 -> row M=lane-16, K 16..31.
// ACT: 0 = none, 1 = relu.
// ---------------------------------------------------------------------------
template <int NT, int ACT>
__global__ __launch_bounds__(256) void gemm_wmma_kernel(
    const float* __restrict__ A, int lda, int acol,
    const bf16_t* __restrict__ Bp, const float* __restrict__ bias,
    float* __restrict__ out, int ldo, int ocol, int M, int K) {
  const int wave = threadIdx.x >> 5;
  const int lane = threadIdx.x & 31;
  const int row0 = (blockIdx.x * 8 + wave) * 16;
  const int Kt = K >> 5;                     // <= 4

  // ---- stage packed B into LDS (whole matrix, fragment-contiguous) --------
  __shared__ __align__(16) bf16_t Bs[NT * 4 * 512];
  {
    const uint4* sv = (const uint4*)Bp;
    uint4* dv = (uint4*)Bs;
    const int n16 = NT * Kt * 64;            // 16B chunks
    for (int i = threadIdx.x; i < n16; i += 256) dv[i] = sv[i];
  }
  __syncthreads();
  if (row0 >= M) return;                     // wave-uniform exit (after barrier)

  int rr0 = row0 + (lane & 15);
  const int r = (rr0 < M) ? rr0 : (M - 1);   // clamp (keeps wave converged)
  const int kb = (lane >> 4) << 4;           // 0 or 16
  const float* abase = A + (size_t)r * lda + acol + kb;

  v8f acc[NT] = {};
  for (int kt = 0; kt < Kt; ++kt) {
    const float* arow = abase + kt * 32;
    if (kt + 1 < Kt) __builtin_prefetch(arow + 32, 0, 0);
    v16bf a;
#pragma unroll
    for (int j = 0; j < 8; ++j) {
      a[j]     = f2bf(arow[j]);
      a[j + 8] = f2bf(arow[8 + j]);
    }
    v16bf bfrag[NT];
#pragma unroll
    for (int n = 0; n < NT; ++n)
      bfrag[n] = *(const v16bf*)(Bs + ((((size_t)n * Kt + kt) * 32 + lane) << 4));
#pragma unroll
    for (int n = 0; n < NT; ++n)
      acc[n] = __builtin_amdgcn_wmma_f32_16x16x32_bf16(
          false, a, false, bfrag[n], (short)0, acc[n], false, false);
  }
  // D layout: element e -> row row0 + e + (lane>=16 ? 8 : 0), col = n*16+(lane&15)
  const int cb = (lane & 15);
  const int rb = row0 + ((lane >> 4) << 3);
#pragma unroll
  for (int n = 0; n < NT; ++n) {
    const int coln = n * 16 + cb;
    const float bv = bias ? bias[coln] : 0.0f;
#pragma unroll
    for (int e = 0; e < 8; ++e) {
      const int rw = rb + e;
      if (rw < M) {
        float v = acc[n][e] + bv;
        if (ACT == 1) v = fmaxf(v, 0.0f);
        out[(size_t)rw * ldo + ocol + coln] = v;
      }
    }
  }
}

// ---------------------------------------------------------------------------
// Elementwise / utility kernels
// ---------------------------------------------------------------------------
__global__ void fill_f32_kernel(float* p, float v, size_t n) {
  size_t i = (size_t)blockIdx.x * blockDim.x + threadIdx.x;
  if (i < n) p[i] = v;
}
__global__ void fill_u32_kernel(unsigned* p, unsigned v, size_t n) {
  size_t i = (size_t)blockIdx.x * blockDim.x + threadIdx.x;
  if (i < n) p[i] = v;
}
__global__ void gelu_inplace_kernel(float* x, size_t n) {
  size_t i = (size_t)blockIdx.x * blockDim.x + threadIdx.x;
  if (i < n) x[i] = gelu_exact(x[i]);
}
// hv = gelu(sk*o + (1-sk)*x) + x   (skip blend then residual)
__global__ void blend_kernel(const float* __restrict__ o, const float* __restrict__ x,
                             const float* __restrict__ skip, float* __restrict__ hv,
                             size_t n) {
  size_t i = (size_t)blockIdx.x * blockDim.x + threadIdx.x;
  if (i >= n) return;
  float sk = 1.0f / (1.0f + expf(-skip[0]));
  float t = sk * o[i] + (1.0f - sk) * x[i];
  hv[i] = gelu_exact(t) + x[i];
}
__global__ void copy_kernel(const float* __restrict__ a, float* __restrict__ b, size_t n) {
  size_t i = (size_t)blockIdx.x * blockDim.x + threadIdx.x;
  if (i < n) b[i] = a[i];
}

// ---------------------------------------------------------------------------
// BatchNorm: per-channel (128) mean/var over rows, then normalize.
// ---------------------------------------------------------------------------
__global__ void bn_stats_kernel(const float* __restrict__ x, int N, float* sums) {
  int col = threadIdx.x;  // 128 threads
  int r0 = blockIdx.x * 256;
  int r1 = r0 + 256;
  if (r1 > N) r1 = N;
  float s = 0.0f, ss = 0.0f;
  for (int r = r0; r < r1; ++r) {
    float v = x[(size_t)r * CDIM + col];
    s += v;
    ss += v * v;
  }
  atomicAdd(&sums[col], s);
  atomicAdd(&sums[CDIM + col], ss);
}
__global__ void bn_apply_kernel(const float* __restrict__ hv, const float* __restrict__ sums,
                                const float* __restrict__ g, const float* __restrict__ b,
                                float* __restrict__ out, int N) {
  size_t i = (size_t)blockIdx.x * blockDim.x + threadIdx.x;
  if (i >= (size_t)N * CDIM) return;
  int col = (int)(i & (CDIM - 1));
  float inv = 1.0f / (float)N;
  float mu = sums[col] * inv;
  float var = sums[CDIM + col] * inv - mu * mu;
  out[i] = (hv[i] - mu) * rsqrtf(var + BN_EPS) * g[col] + b[col];
}

// ---------------------------------------------------------------------------
// Edge-phase kernels (one wave per edge: coalesced 128-wide row accesses;
// lane l covers element h*32+l, i.e. exactly one element per head)
// ---------------------------------------------------------------------------
// Pass 1: alpha[e,h] = dot(q[dst],krel[src])_h * p_rel[h]/sqrt(D); atomicMax.
__global__ __launch_bounds__(256) void edge_alpha_kernel(
    const int* __restrict__ src, const int* __restrict__ dst,
    const float* __restrict__ q, const float* __restrict__ krel,
    const float* __restrict__ p_rel, float* __restrict__ alpha,
    unsigned* __restrict__ amax, int E) {
  int wid = (int)((blockIdx.x * (size_t)blockDim.x + threadIdx.x) >> 5);
  int lane = threadIdx.x & 31;
  if (wid >= E) return;
  int s = src[wid], d = dst[wid];
  const float* qd = q + (size_t)d * CDIM;
  const float* ks = krel + (size_t)s * CDIM;
  float p[HDIM];
#pragma unroll
  for (int h = 0; h < HDIM; ++h)
    p[h] = qd[h * DDIM + lane] * ks[h * DDIM + lane];
#pragma unroll
  for (int off = 16; off > 0; off >>= 1) {
#pragma unroll
    for (int h = 0; h < HDIM; ++h)
      p[h] += __shfl_xor(p[h], off, 32);
  }
  if (lane < HDIM) {
    float a = p[lane] * p_rel[lane] * 0.17677669529663687f;  // 1/sqrt(32)
    alpha[(size_t)wid * HDIM + lane] = a;
    atomicMax(&amax[(size_t)d * HDIM + lane], f2ord(a));
  }
}
// Pass 2: alpha <- exp(alpha - amax[dst]); atomicAdd asum[dst].
__global__ void edge_exp_kernel(const int* __restrict__ dst, float* __restrict__ alpha,
                                const unsigned* __restrict__ amax, float* __restrict__ asum,
                                int E) {
  int t = blockIdx.x * blockDim.x + threadIdx.x;
  if (t >= E * HDIM) return;
  int e = t >> 2, h = t & 3;
  int d = dst[e];
  float m = ord2f(amax[(size_t)d * HDIM + h]);
  float w = expf(alpha[t] - m);
  alpha[t] = w;
  atomicAdd(&asum[(size_t)d * HDIM + h], w);
}
// Pass 3: agg[dst] += vrel[src] * alpha/(asum[dst]+eps); wave per edge,
// 4 coalesced f32 atomics per lane.
__global__ __launch_bounds__(256) void edge_scatter_kernel(
    const int* __restrict__ src, const int* __restrict__ dst,
    const float* __restrict__ alpha, const float* __restrict__ asum,
    const float* __restrict__ vrel, float* __restrict__ agg, int E) {
  int wid = (int)((blockIdx.x * (size_t)blockDim.x + threadIdx.x) >> 5);
  int lane = threadIdx.x & 31;
  if (wid >= E) return;
  int s = src[wid], d = dst[wid];
  float w = 0.0f;
  if (lane < HDIM)
    w = alpha[(size_t)wid * HDIM + lane] /
        (asum[(size_t)d * HDIM + lane] + 1e-16f);
  float wh[HDIM];
#pragma unroll
  for (int h = 0; h < HDIM; ++h) wh[h] = __shfl(w, h, 32);
  const float* vs = vrel + (size_t)s * CDIM;
  float* ag = agg + (size_t)d * CDIM;
#pragma unroll
  for (int h = 0; h < HDIM; ++h)
    atomicAdd(&ag[h * DDIM + lane], vs[h * DDIM + lane] * wh[h]);
}

// ---------------------------------------------------------------------------
// Host-side orchestration
// ---------------------------------------------------------------------------
static void launch_gemm(const float* A, int lda, int acol, const bf16_t* Bp,
                        const float* bias, float* out, int ldo, int ocol,
                        int M, int K, int N, int act, hipStream_t s) {
  dim3 blk(256, 1, 1), grd((unsigned)((M + 127) / 128), 1, 1);
  if (N == 128) {
    if (act == 1)
      gemm_wmma_kernel<8, 1><<<grd, blk, 0, s>>>(A, lda, acol, Bp, bias, out, ldo, ocol, M, K);
    else
      gemm_wmma_kernel<8, 0><<<grd, blk, 0, s>>>(A, lda, acol, Bp, bias, out, ldo, ocol, M, K);
  } else {
    gemm_wmma_kernel<2, 0><<<grd, blk, 0, s>>>(A, lda, acol, Bp, bias, out, ldo, ocol, M, K);
  }
}

extern "C" void kernel_launch(void* const* d_in, const int* in_sizes, int n_in,
                              void* d_out, int out_size, void* d_ws, size_t ws_size,
                              hipStream_t stream) {
  (void)n_in; (void)out_size; (void)ws_size;
  // --- inputs (insertion-order pytree flatten of setup_inputs()) -----------
  // 0 x_author, 1 x_paper, 2 edge_writes, 3 edge_rev, 4 edge_cites,
  // 5 lin.author.W, 6 lin.author.b, 7 lin.paper.W, 8 lin.paper.b,
  // conv l at base 9+27l:
  //   +0..+11 kqv (author k/q/v W,b then paper), +12..+15 alin (author W,b; paper W,b),
  //   +16 skip.author, +17 skip.paper, +18..+26 rel {a_rel,m_rel,p_rel} x 3 edge types
  // bn at 63: per layer {author g,b, paper g,b}
  const float* xa0 = (const float*)d_in[0];
  const float* xp0 = (const float*)d_in[1];
  const int NA = in_sizes[0] / CDIM;
  const int NP = in_sizes[1] / CDIM;
  const int Earr[3] = { in_sizes[2] / 2, in_sizes[3] / 2, in_sizes[4] / 2 };
  const int* eptr[3] = { (const int*)d_in[2], (const int*)d_in[3], (const int*)d_in[4] };
  const int srcT[3] = { 0, 1, 1 };   // author->paper, paper->author, paper->paper
  const int dstT[3] = { 1, 0, 1 };
  int EMAX = Earr[0]; for (int t = 1; t < 3; ++t) if (Earr[t] > EMAX) EMAX = Earr[t];
  const int NDMAX = (NA > NP) ? NA : NP;

  auto F = [&](int i) -> const float* { return (const float*)d_in[i]; };

  // --- workspace carve-up ---------------------------------------------------
  size_t off = 0;
  auto alloc_raw = [&](size_t bytes) -> void* {
    void* p = (void*)((char*)d_ws + off);
    off += (bytes + 255) & ~(size_t)255;
    return p;
  };
  auto alloc_f = [&](size_t elems) -> float* { return (float*)alloc_raw(elems * 4); };
  auto alloc_b = [&](size_t elems) -> bf16_t* { return (bf16_t*)alloc_raw(elems * 2); };

  float* xa = alloc_f((size_t)NA * CDIM);
  float* xp = alloc_f((size_t)NP * CDIM);
  float* k_[2] = { alloc_f((size_t)NA * CDIM), alloc_f((size_t)NP * CDIM) };
  float* q_[2] = { alloc_f((size_t)NA * CDIM), alloc_f((size_t)NP * CDIM) };
  float* v_[2] = { alloc_f((size_t)NA * CDIM), alloc_f((size_t)NP * CDIM) };
  float* krel = alloc_f((size_t)NDMAX * CDIM);
  float* vrel = alloc_f((size_t)NDMAX * CDIM);
  float* agg_[2] = { alloc_f((size_t)NA * CDIM), alloc_f((size_t)NP * CDIM) };
  float* o_[2]   = { alloc_f((size_t)NA * CDIM), alloc_f((size_t)NP * CDIM) };
  float* alpha = alloc_f((size_t)EMAX * HDIM);
  float* asum  = alloc_f((size_t)NDMAX * HDIM);
  unsigned* amax = (unsigned*)alloc_raw((size_t)NDMAX * HDIM * 4);
  float* bnsum = alloc_f(2 * CDIM);

  const size_t PK128 = 128 * 128;  // packed bf16 elements for 128x128 W
  const size_t PK32  = 32 * 32;
  bf16_t* pk_lin[2];
  bf16_t* pk_kqv[2][2][3];
  bf16_t* pk_alin[2][2];
  bf16_t* pk_rel[2][3][2][HDIM];
  for (int nt = 0; nt < 2; ++nt) pk_lin[nt] = alloc_b(PK128);
  for (int l = 0; l < 2; ++l) {
    for (int nt = 0; nt < 2; ++nt)
      for (int j = 0; j < 3; ++j) pk_kqv[l][nt][j] = alloc_b(PK128);
    for (int nt = 0; nt < 2; ++nt) pk_alin[l][nt] = alloc_b(PK128);
    for (int t = 0; t < 3; ++t)
      for (int am = 0; am < 2; ++am)
        for (int h = 0; h < HDIM; ++h) pk_rel[l][t][am][h] = alloc_b(PK32);
  }

  // --- weight packing (f32 -> bf16 fragments) -------------------------------
  auto pack = [&](const float* W, bf16_t* dst, int K, int N) {
    int total = (N >> 4) * (K >> 5) * 32;
    pack_b_kernel<<<(total + 63) / 64, 64, 0, stream>>>(W, dst, K, N);
  };
  pack(F(5), pk_lin[0], 128, 128);
  pack(F(7), pk_lin[1], 128, 128);
  for (int l = 0; l < 2; ++l) {
    int cb = 9 + l * 27;
    for (int nt = 0; nt < 2; ++nt)
      for (int j = 0; j < 3; ++j)
        pack(F(cb + nt * 6 + j * 2), pk_kqv[l][nt][j], 128, 128);
    pack(F(cb + 12), pk_alin[l][0], 128, 128);
    pack(F(cb + 14), pk_alin[l][1], 128, 128);
    for (int t = 0; t < 3; ++t)
      for (int am = 0; am < 2; ++am)
        for (int h = 0; h < HDIM; ++h)
          pack(F(cb + 18 + t * 3 + am) + (size_t)h * DDIM * DDIM,
               pk_rel[l][t][am][h], 32, 32);
  }

  // --- input linear + relu --------------------------------------------------
  launch_gemm(xa0, CDIM, 0, pk_lin[0], F(6), xa, CDIM, 0, NA, 128, 128, 1, stream);
  launch_gemm(xp0, CDIM, 0, pk_lin[1], F(8), xp, CDIM, 0, NP, 128, 128, 1, stream);

  float* x_[2] = { xa, xp };
  int    N_[2] = { NA, NP };
  const unsigned ORD_NEG_INF = 0x007FFFFFu;  // f2ord(-inf)

  // --- layers ---------------------------------------------------------------
  for (int l = 0; l < 2; ++l) {
    int cb = 9 + l * 27;
    // K/Q/V projections per node type (WMMA GEMMs)
    for (int nt = 0; nt < 2; ++nt) {
      int b = cb + nt * 6;
      launch_gemm(x_[nt], CDIM, 0, pk_kqv[l][nt][0], F(b + 1), k_[nt], CDIM, 0, N_[nt], 128, 128, 0, stream);
      launch_gemm(x_[nt], CDIM, 0, pk_kqv[l][nt][1], F(b + 3), q_[nt], CDIM, 0, N_[nt], 128, 128, 0, stream);
      launch_gemm(x_[nt], CDIM, 0, pk_kqv[l][nt][2], F(b + 5), v_[nt], CDIM, 0, N_[nt], 128, 128, 0, stream);
    }
    // zero aggregates
    for (int nt = 0; nt < 2; ++nt) {
      size_t n = (size_t)N_[nt] * CDIM;
      fill_f32_kernel<<<(unsigned)((n + 255) / 256), 256, 0, stream>>>(agg_[nt], 0.0f, n);
    }
    // edge types
    for (int t = 0; t < 3; ++t) {
      int st = srcT[t], dt = dstT[t];
      int E = Earr[t];
      const int* src = eptr[t];
      const int* dst = eptr[t] + E;
      // relation transforms: per-head 32x32 WMMA GEMMs on source K/V
      for (int h = 0; h < HDIM; ++h) {
        launch_gemm(k_[st], CDIM, h * DDIM, pk_rel[l][t][0][h], nullptr,
                    krel, CDIM, h * DDIM, N_[st], 32, 32, 0, stream);
        launch_gemm(v_[st], CDIM, h * DDIM, pk_rel[l][t][1][h], nullptr,
                    vrel, CDIM, h * DDIM, N_[st], 32, 32, 0, stream);
      }
      size_t nd = (size_t)N_[dt] * HDIM;
      fill_u32_kernel<<<(unsigned)((nd + 255) / 256), 256, 0, stream>>>(amax, ORD_NEG_INF, nd);
      fill_f32_kernel<<<(unsigned)((nd + 255) / 256), 256, 0, stream>>>(asum, 0.0f, nd);
      // one wave per edge
      unsigned ewaves = (unsigned)(((size_t)E * 32 + 255) / 256);
      edge_alpha_kernel<<<ewaves, 256, 0, stream>>>(
          src, dst, q_[dt], krel, F(cb + 18 + t * 3 + 2), alpha, amax, E);
      edge_exp_kernel<<<(E * HDIM + 255) / 256, 256, 0, stream>>>(dst, alpha, amax, asum, E);
      edge_scatter_kernel<<<ewaves, 256, 0, stream>>>(
          src, dst, alpha, asum, vrel, agg_[dt], E);
    }
    // epilogue per node type: gelu(agg) @ alin + skip blend + residual + BN
    for (int nt = 0; nt < 2; ++nt) {
      size_t n = (size_t)N_[nt] * CDIM;
      gelu_inplace_kernel<<<(unsigned)((n + 255) / 256), 256, 0, stream>>>(agg_[nt], n);
      launch_gemm(agg_[nt], CDIM, 0, pk_alin[l][nt], F(cb + 12 + nt * 2 + 1),
                  o_[nt], CDIM, 0, N_[nt], 128, 128, 0, stream);
      blend_kernel<<<(unsigned)((n + 255) / 256), 256, 0, stream>>>(
          o_[nt], x_[nt], F(cb + 16 + nt), agg_[nt], n);
      fill_f32_kernel<<<1, 256, 0, stream>>>(bnsum, 0.0f, 2 * CDIM);
      bn_stats_kernel<<<(N_[nt] + 255) / 256, CDIM, 0, stream>>>(agg_[nt], N_[nt], bnsum);
      bn_apply_kernel<<<(unsigned)((n + 255) / 256), 256, 0, stream>>>(
          agg_[nt], bnsum, F(63 + l * 4 + nt * 2), F(63 + l * 4 + nt * 2 + 1),
          x_[nt], N_[nt]);
    }
  }

  // --- outputs: (x_author, x_paper) concatenated ---------------------------
  float* out = (float*)d_out;
  size_t na = (size_t)NA * CDIM, np = (size_t)NP * CDIM;
  copy_kernel<<<(unsigned)((na + 255) / 256), 256, 0, stream>>>(xa, out, na);
  copy_kernel<<<(unsigned)((np + 255) / 256), 256, 0, stream>>>(xp, out + na, np);
}
